// NMFormer_62466004353049
// MI455X (gfx1250) — matrix operand order repaired and checked by reading, not compiled, not run on hardware
//
#include <hip/hip_runtime.h>

// ---- problem constants ----
constexpr int DIM = 512;   // D
constexpr int VN  = 256;   // videos
constexpr int FN  = 64;    // frames
constexpr int TN  = 512;   // text tokens
constexpr int NITERS = 10;
#define LAMBC 0.15f
#define EPSC 1e-6f
#define LN_EPSC 1e-5f
#define ETA (LAMBC / (float)NITERS)

typedef __attribute__((ext_vector_type(16))) __bf16 v16bf;
typedef __attribute__((ext_vector_type(16))) short  v16s;
typedef __attribute__((ext_vector_type(8)))  float  v8f;
typedef __attribute__((ext_vector_type(2)))  float  v2f;

union BFU { v16s s; v16bf bf; };

__device__ __forceinline__ unsigned short f2bf(float f) {
  unsigned int u = __float_as_uint(f);
  u += 0x7FFFu + ((u >> 16) & 1u);           // round-to-nearest-even
  return (unsigned short)(u >> 16);
}

// A fragment (bf16 16x32): lane holds row M=lane&15; lanes<16: K 0..15,
// lanes>=16: K 16..31 (contiguous, matching the B fragment's K order).
__device__ __forceinline__ v16bf load_afrag_bf16(const unsigned short* base, int ld,
                                                 int row0, int k0, int lane) {
  int m = lane & 15;
  int koff = (lane < 16) ? 0 : 16;
  const unsigned short* p = base + (size_t)(row0 + m) * ld + k0 + koff;
  BFU u;
#pragma unroll
  for (int i = 0; i < 8; ++i) u.s[i] = (short)p[i];
#pragma unroll
  for (int i = 0; i < 8; ++i) u.s[8 + i] = (short)p[8 + i];
  return u.bf;
}

// B fragment (bf16 32x16): lane = col N + 16*(K>=16); source is "row per N",
// K contiguous within the row (so B reads are contiguous 32B per lane).
__device__ __forceinline__ v16bf load_bfrag_bf16(const unsigned short* base, int ld,
                                                 int n0, int k0, int lane) {
  int n = lane & 15;
  int koff = (lane < 16) ? 0 : 16;
  const unsigned short* p = base + (size_t)(n0 + n) * ld + k0 + koff;
  BFU u;
#pragma unroll
  for (int i = 0; i < 16; ++i) u.s[i] = (short)p[i];
  return u.bf;
}

// f32 WMMA 16x16x4 fragments
__device__ __forceinline__ v2f load_afrag_f32(const float* base, int ld,
                                              int row0, int k0, int lane) {
  int m = lane & 15;
  int koff = (lane < 16) ? 0 : 2;
  const float* p = base + (row0 + m) * ld + k0 + koff;
  v2f r; r[0] = p[0]; r[1] = p[1];
  return r;
}
__device__ __forceinline__ v2f load_bfrag_f32(const float* base, int ld,
                                              int n0, int k0, int lane) {
  int n = lane & 15;
  int koff = (lane < 16) ? 0 : 2;
  const float* p = base + (n0 + n) * ld + k0 + koff;
  v2f r; r[0] = p[0]; r[1] = p[1];
  return r;
}

// ---------- LayerNorm rows -> bf16 (optionally also transposed copy) ----------
__global__ void ln_rows_kernel(const float* __restrict__ x, const float* __restrict__ w,
                               const float* __restrict__ b, unsigned short* __restrict__ out_bf,
                               unsigned short* __restrict__ outT_bf, int nrows) {
  int wave = threadIdx.x >> 5;
  int lane = threadIdx.x & 31;
  int row = blockIdx.x * 8 + wave;
  if (row >= nrows) return;
  const float* src = x + (size_t)row * DIM;
  float vals[16];
#pragma unroll
  for (int j = 0; j < 16; ++j) vals[j] = src[lane + 32 * j];
  float s = 0.f;
#pragma unroll
  for (int j = 0; j < 16; ++j) s += vals[j];
#pragma unroll
  for (int m = 1; m <= 16; m <<= 1) s += __shfl_xor(s, m, 32);
  float mean = s * (1.0f / DIM);
  float s2 = 0.f;
#pragma unroll
  for (int j = 0; j < 16; ++j) { float d = vals[j] - mean; s2 += d * d; }
#pragma unroll
  for (int m = 1; m <= 16; m <<= 1) s2 += __shfl_xor(s2, m, 32);
  float rstd = rsqrtf(s2 * (1.0f / DIM) + LN_EPSC);
#pragma unroll
  for (int j = 0; j < 16; ++j) {
    int col = lane + 32 * j;
    float y = (vals[j] - mean) * rstd * w[col] + b[col];
    unsigned short h = f2bf(y);
    out_bf[(size_t)row * DIM + col] = h;
    if (outT_bf) {  // vdT[v][d][f]
      int v = row >> 6, f = row & 63;
      outT_bf[((size_t)v * DIM + col) * FN + f] = h;
    }
  }
}

__global__ void cvt_bf16_kernel(const float* __restrict__ x, unsigned short* __restrict__ y, int n) {
  int i = blockIdx.x * 256 + threadIdx.x;
  if (i < n) y[i] = f2bf(x[i]);
}

// ---------- gram[v] = vd[v] @ vd[v]^T  (64x64, K=512) ----------
__global__ void __launch_bounds__(256) gram_kernel(const unsigned short* __restrict__ vd,
                                                   float* __restrict__ gram) {
  int v = blockIdx.x;
  int wave = threadIdx.x >> 5, lane = threadIdx.x & 31;
  int lh = lane >> 4, ln15 = lane & 15;
  const unsigned short* base = vd + (size_t)v * FN * DIM;
  float* gout = gram + (size_t)v * FN * FN;
  int mi = wave >> 1;
  int nb = (wave & 1) * 2;
  v8f acc[2];
#pragma unroll
  for (int s = 0; s < 2; ++s) { v8f z = {}; acc[s] = z; }
  for (int kc = 0; kc < 16; ++kc) {
    v16bf af = load_afrag_bf16(base, DIM, mi * 16, kc * 32, lane);   // shared by both subtiles
#pragma unroll
    for (int s = 0; s < 2; ++s) {
      v16bf bf = load_bfrag_bf16(base, DIM, (nb + s) * 16, kc * 32, lane);
      acc[s] = __builtin_amdgcn_wmma_f32_16x16x32_bf16(false, af, false, bf, (short)0, acc[s], false, false);
    }
  }
#pragma unroll
  for (int s = 0; s < 2; ++s) {
    int col = (nb + s) * 16 + ln15;
#pragma unroll
    for (int r = 0; r < 8; ++r)
      gout[(mi * 16 + r + 8 * lh) * FN + col] = acc[s][r];
  }
}

// ---------- fused: sims -> 10 iterations -> recon -> row-normalize -> LN2 ----------
__global__ void __launch_bounds__(256) attn_kernel(const unsigned short* __restrict__ t_bf,
    const unsigned short* __restrict__ vd, const unsigned short* __restrict__ vdT,
    const float* __restrict__ gram, const float* __restrict__ ln2w, const float* __restrict__ ln2b,
    float* __restrict__ out) {
  __shared__ float gram_lds[FN * FN];            // 16 KB
  __shared__ float a_stage[FN * FN];             // 16 KB
  __shared__ unsigned short ab_stage[FN * FN];   // 8 KB
  __shared__ float red1[FN];
  __shared__ float red2[FN];

  int tt = blockIdx.x;          // t-tile 0..7
  int v  = blockIdx.y;          // video 0..255
  int tid = threadIdx.x;
  int wave = tid >> 5, lane = tid & 31;
  int lh = lane >> 4, ln15 = lane & 15;

  // Async DMA the 16KB gram tile into LDS (ASYNCcnt path): 256 lanes x 4 x b128
  {
    const float* gsrc = gram + (size_t)v * FN * FN + tid * 4;
    unsigned ldsoff = (unsigned)(uintptr_t)gram_lds + (unsigned)tid * 16u;
#pragma unroll
    for (int i = 0; i < 4; ++i) {
      asm volatile("global_load_async_to_lds_b128 %0, %1, off"
                   :: "v"(ldsoff + (unsigned)i * 4096u), "v"(gsrc + i * 1024)
                   : "memory");
    }
  }
  if (tid < FN) { red1[tid] = 0.f; red2[tid] = 0.f; }
  asm volatile("s_wait_asynccnt 0x0" ::: "memory");
  __syncthreads();

  int mi = wave >> 1;           // 16-row block of the 64-row tile
  int nb = (wave & 1) * 2;      // pair of F-subtiles
  int t0 = tt * 64;
  const unsigned short* vbase = vd + (size_t)v * FN * DIM;

  // sims tile (a := sims), bf16 WMMA over K=512; A fragment shared by both subtiles
  v8f a_reg[2], s_reg[2];
#pragma unroll
  for (int s = 0; s < 2; ++s) { v8f z = {}; a_reg[s] = z; }
  for (int kc = 0; kc < 16; ++kc) {
    v16bf af = load_afrag_bf16(t_bf, DIM, t0 + mi * 16, kc * 32, lane);
#pragma unroll
    for (int s = 0; s < 2; ++s) {
      v16bf bf = load_bfrag_bf16(vbase, DIM, (nb + s) * 16, kc * 32, lane);
      a_reg[s] = __builtin_amdgcn_wmma_f32_16x16x32_bf16(false, af, false, bf, (short)0, a_reg[s], false, false);
    }
  }
#pragma unroll
  for (int s = 0; s < 2; ++s) {
    s_reg[s] = a_reg[s];
#pragma unroll
    for (int r = 0; r < 8; ++r)
      a_stage[(mi * 16 + r + 8 * lh) * FN + (nb + s) * 16 + ln15] = a_reg[s][r];
  }
  __syncthreads();

  // 10 on-chip iterations: dnm = a @ gram (symmetric) via exact f32 WMMA
  for (int it = 0; it < NITERS; ++it) {
    v8f dn[2];
#pragma unroll
    for (int s = 0; s < 2; ++s) { v8f z = {}; dn[s] = z; }
    for (int nc = 0; nc < 16; ++nc) {
      v2f af = load_afrag_f32(a_stage, FN, mi * 16, nc * 4, lane);   // shared by both subtiles
#pragma unroll
      for (int s = 0; s < 2; ++s) {
        v2f bf = load_bfrag_f32(gram_lds, FN, (nb + s) * 16, nc * 4, lane);
        dn[s] = __builtin_amdgcn_wmma_f32_16x16x4_f32(false, af, false, bf, (short)0, dn[s], false, false);
      }
    }
    __syncthreads();
#pragma unroll
    for (int s = 0; s < 2; ++s) {
#pragma unroll
      for (int r = 0; r < 8; ++r) {
        float a = a_reg[s][r] - ETA * (dn[s][r] + EPSC - s_reg[s][r]);
        a = (a <= 0.f) ? EPSC : a;
        a_reg[s][r] = a;
        a_stage[(mi * 16 + r + 8 * lh) * FN + (nb + s) * 16 + ln15] = a;
      }
    }
    __syncthreads();
  }

  // attn_tilde -> bf16 staging for recon GEMM
#pragma unroll
  for (int s = 0; s < 2; ++s)
#pragma unroll
    for (int r = 0; r < 8; ++r)
      ab_stage[(mi * 16 + r + 8 * lh) * FN + (nb + s) * 16 + ln15] = f2bf(a_reg[s][r]);
  __syncthreads();

  // recon = attn_tilde @ vd : rows mi*16..+15, wave half h covers 256 D-cols.
  // A fragments (K=64 -> two chunks) are invariant across all 16 N-subtiles: hoist.
  int h = wave & 1;
  const unsigned short* vtbase = vdT + (size_t)v * DIM * FN;
  v16bf af0 = load_afrag_bf16(ab_stage, FN, mi * 16, 0, lane);
  v16bf af1 = load_afrag_bf16(ab_stage, FN, mi * 16, 32, lane);
  v8f rc[16];
  float s1p[8], s2p[8];
#pragma unroll
  for (int r = 0; r < 8; ++r) { s1p[r] = 0.f; s2p[r] = 0.f; }
  for (int ns = 0; ns < 16; ++ns) {
    v8f acc = {};
    v16bf bf0 = load_bfrag_bf16(vtbase, FN, h * 256 + ns * 16, 0, lane);
    v16bf bf1 = load_bfrag_bf16(vtbase, FN, h * 256 + ns * 16, 32, lane);
    acc = __builtin_amdgcn_wmma_f32_16x16x32_bf16(false, af0, false, bf0, (short)0, acc, false, false);
    acc = __builtin_amdgcn_wmma_f32_16x16x32_bf16(false, af1, false, bf1, (short)0, acc, false, false);
    rc[ns] = acc;
#pragma unroll
    for (int r = 0; r < 8; ++r) { float x = acc[r]; s1p[r] += x; s2p[r] += x * x; }
  }
  // reduce across the 16 lanes of each half (they share the same 8 rows)
#pragma unroll
  for (int m = 1; m <= 8; m <<= 1) {
#pragma unroll
    for (int r = 0; r < 8; ++r) {
      s1p[r] += __shfl_xor(s1p[r], m, 32);
      s2p[r] += __shfl_xor(s2p[r], m, 32);
    }
  }
  if (ln15 == 0) {
#pragma unroll
    for (int r = 0; r < 8; ++r) {
      atomicAdd(&red1[mi * 16 + r + 8 * lh], s1p[r]);
      atomicAdd(&red2[mi * 16 + r + 8 * lh], s2p[r]);
    }
  }
  __syncthreads();

  // attn_out = recon/||recon||; LN2 with E[x^2] = 1/D exactly
  for (int ns = 0; ns < 16; ++ns) {
    int col = h * 256 + ns * 16 + ln15;
    float w2 = ln2w[col], b2 = ln2b[col];
#pragma unroll
    for (int r = 0; r < 8; ++r) {
      int rloc = mi * 16 + r + 8 * lh;
      float nrm = sqrtf(red2[rloc]) + 1e-20f;
      float mean = red1[rloc] / (nrm * (float)DIM);
      float rstd = rsqrtf((1.0f / DIM) - mean * mean + LN_EPSC);
      float xx = rc[ns][r] / nrm;
      out[((size_t)v * TN + (t0 + rloc)) * DIM + col] = (xx - mean) * rstd * w2 + b2;
    }
  }
}

// ---------- final: linear (bf16 WMMA, K=512) + residual + LN3, in-place on out ----------
__global__ void __launch_bounds__(256) final_kernel(const unsigned short* __restrict__ wbf,
    const float* __restrict__ pb, const float* __restrict__ ln3w, const float* __restrict__ ln3b,
    float* __restrict__ out) {
  __shared__ unsigned short a_lds[32 * DIM];   // 32 KB bf16 A block
  __shared__ float red1[32];
  __shared__ float red2[32];
  int tid = threadIdx.x;
  int wave = tid >> 5, lane = tid & 31;
  int lh = lane >> 4, ln15 = lane & 15;
  size_t r0 = (size_t)blockIdx.x * 32;

  for (int i = tid; i < 32 * DIM; i += 256) a_lds[i] = f2bf(out[r0 * DIM + i]);
  if (tid < 32) { red1[tid] = 0.f; red2[tid] = 0.f; }
  __syncthreads();

  int mi = wave & 1;             // 16-row block
  int nbase = (wave >> 1) * 8;   // 8 output-col subtiles per wave
  v8f acc[8];
#pragma unroll
  for (int s = 0; s < 8; ++s) { v8f z = {}; acc[s] = z; }
  for (int kc = 0; kc < 16; ++kc) {
    v16bf af = load_afrag_bf16(a_lds, DIM, mi * 16, kc * 32, lane);  // one A per K chunk
#pragma unroll
    for (int s = 0; s < 8; ++s) {
      v16bf bf = load_bfrag_bf16(wbf, DIM, (nbase + s) * 16, kc * 32, lane);
      acc[s] = __builtin_amdgcn_wmma_f32_16x16x32_bf16(false, af, false, bf, (short)0, acc[s], false, false);
    }
  }

  float s1p[8], s2p[8];
#pragma unroll
  for (int r = 0; r < 8; ++r) { s1p[r] = 0.f; s2p[r] = 0.f; }
#pragma unroll
  for (int s = 0; s < 8; ++s) {
    int col = (nbase + s) * 16 + ln15;
    float bias = pb[col];
#pragma unroll
    for (int r = 0; r < 8; ++r) {
      int rloc = mi * 16 + r + 8 * lh;
      float y = out[(r0 + rloc) * DIM + col] + acc[s][r] + bias;  // residual + linear + bias
      acc[s][r] = y;
      s1p[r] += y; s2p[r] += y * y;
    }
  }
#pragma unroll
  for (int m = 1; m <= 8; m <<= 1) {
#pragma unroll
    for (int r = 0; r < 8; ++r) {
      s1p[r] += __shfl_xor(s1p[r], m, 32);
      s2p[r] += __shfl_xor(s2p[r], m, 32);
    }
  }
  if (ln15 == 0) {
#pragma unroll
    for (int r = 0; r < 8; ++r) {
      atomicAdd(&red1[mi * 16 + r + 8 * lh], s1p[r]);
      atomicAdd(&red2[mi * 16 + r + 8 * lh], s2p[r]);
    }
  }
  __syncthreads();

#pragma unroll
  for (int s = 0; s < 8; ++s) {
    int col = (nbase + s) * 16 + ln15;
    float w3 = ln3w[col], b3 = ln3b[col];
#pragma unroll
    for (int r = 0; r < 8; ++r) {
      int rloc = mi * 16 + r + 8 * lh;
      float mean = red1[rloc] * (1.0f / DIM);
      float var = red2[rloc] * (1.0f / DIM) - mean * mean;
      float rstd = rsqrtf(var + LN_EPSC);
      out[(r0 + rloc) * DIM + col] = (acc[s][r] - mean) * rstd * w3 + b3;
    }
  }
}

extern "C" void kernel_launch(void* const* d_in, const int* in_sizes, int n_in,
                              void* d_out, int out_size, void* d_ws, size_t ws_size,
                              hipStream_t stream) {
  (void)in_sizes; (void)n_in; (void)out_size; (void)ws_size;
  const float* text = (const float*)d_in[0];
  const float* video = (const float*)d_in[1];
  const float* ln1w = (const float*)d_in[2];
  const float* ln1b = (const float*)d_in[3];
  const float* ln2w = (const float*)d_in[4];
  const float* ln2b = (const float*)d_in[5];
  const float* ln3w = (const float*)d_in[6];
  const float* ln3b = (const float*)d_in[7];
  const float* pw   = (const float*)d_in[8];
  const float* pb   = (const float*)d_in[9];
  float* out = (float*)d_out;

  char* ws = (char*)d_ws;
  size_t off = 0;
  auto alloc = [&](size_t bytes) -> char* {
    char* p = ws + off;
    off = (off + bytes + 255) & ~(size_t)255;
    return p;
  };
  unsigned short* t_bf   = (unsigned short*)alloc((size_t)TN * DIM * 2);          // 0.5 MB
  unsigned short* vd_bf  = (unsigned short*)alloc((size_t)VN * FN * DIM * 2);     // 16 MB
  unsigned short* vdT_bf = (unsigned short*)alloc((size_t)VN * DIM * FN * 2);     // 16 MB
  float*          gram   = (float*)alloc((size_t)VN * FN * FN * 4);               // 4 MB
  unsigned short* w_bf   = (unsigned short*)alloc((size_t)DIM * DIM * 2);         // 0.5 MB

  ln_rows_kernel<<<TN / 8, 256, 0, stream>>>(text, ln1w, ln1b, t_bf, nullptr, TN);
  ln_rows_kernel<<<(VN * FN) / 8, 256, 0, stream>>>(video, ln1w, ln1b, vd_bf, vdT_bf, VN * FN);
  cvt_bf16_kernel<<<(DIM * DIM) / 256, 256, 0, stream>>>(pw, w_bf, DIM * DIM);
  gram_kernel<<<VN, 256, 0, stream>>>(vd_bf, gram);
  attn_kernel<<<dim3(TN / 64, VN), 256, 0, stream>>>(t_bf, vd_bf, vdT_bf, gram, ln2w, ln2b, out);
  final_kernel<<<(VN * TN) / 32, 256, 0, stream>>>(w_bf, pb, ln3w, ln3b, out);
}
